// QwenImageTransformerBlock_16071767622423
// MI455X (gfx1250) — matrix operand order repaired and checked by reading, not compiled, not run on hardware
//
#include <hip/hip_runtime.h>
#include <hip/hip_bf16.h>
#include <math.h>

// ---------------------------------------------------------------------------
// Qwen-Image dual-stream transformer block, MI455X (gfx1250, wave32, WMMA).
// fp32 in/out; bf16 WMMA (v_wmma_f32_16x16x32_bf16) with fp32 accumulation.
// ---------------------------------------------------------------------------

#define BB     2
#define S_IMG  1024
#define S_TXT  256
#define SS     1280          // S_IMG + S_TXT
#define DD     3072
#define HH     24
#define DH     128
#define FF     12288
#define EPS    1e-6f

typedef __attribute__((ext_vector_type(16))) __bf16 v16bf;
typedef __attribute__((ext_vector_type(8)))  float  v8f;

// ---------------------------------------------------------------------------
// WMMA helpers
// ---------------------------------------------------------------------------
__device__ __forceinline__ v8f wmma_bf16(v16bf a, v16bf b, v8f c) {
  // 8 args: (neg_a, A, neg_b, B, c_mod, C, reuse_a, reuse_b)
  return __builtin_amdgcn_wmma_f32_16x16x32_bf16(false, a, false, b, (short)0, c,
                                                 false, false);
}

// A-matrix fragment, 16x32 (MxK), fp32 source -> bf16.
// Lane l: M = l&15 ; lanes 0-15 hold K=0..15, lanes 16-31 K=16..31.
__device__ __forceinline__ v16bf load_frag_a(const float* A, int lda,
                                             int m_base, int k_base, int lane) {
  int m  = m_base + (lane & 15);
  int kb = ((lane >> 4) & 1) * 16;
  const float* p = A + (size_t)m * lda + k_base + kb;
  const float4* p4 = (const float4*)p;
  float4 x0 = p4[0], x1 = p4[1];          // K = kb .. kb+7
  const float4* q4 = (const float4*)(p + 8);
  float4 y0 = q4[0], y1 = q4[1];          // K = kb+8 .. kb+15
  v16bf f;
  f[0]=(__bf16)x0.x; f[1]=(__bf16)x0.y; f[2]=(__bf16)x0.z; f[3]=(__bf16)x0.w;
  f[4]=(__bf16)x1.x; f[5]=(__bf16)x1.y; f[6]=(__bf16)x1.z; f[7]=(__bf16)x1.w;
  f[8]=(__bf16)y0.x; f[9]=(__bf16)y0.y; f[10]=(__bf16)y0.z; f[11]=(__bf16)y0.w;
  f[12]=(__bf16)y1.x; f[13]=(__bf16)y1.y; f[14]=(__bf16)y1.z; f[15]=(__bf16)y1.w;
  return f;
}

// B-matrix fragment, 32x16 (KxN), from W stored row-major [N,K] (B(k,n)=W(n,k)).
// Lane l: N = l&15 ; lanes 0-15 hold K=0..15, lanes 16-31 K=16..31. Contiguous load.
__device__ __forceinline__ v16bf load_frag_b(const float* W, int ldw,
                                             int n_base, int k_base, int lane) {
  int n   = n_base + (lane & 15);
  int kb2 = k_base + ((lane >> 4) & 1) * 16;
  const float4* p4 = (const float4*)(W + (size_t)n * ldw + kb2);
  v16bf f;
  #pragma unroll
  for (int v = 0; v < 4; ++v) {
    float4 x = p4[v];
    f[4*v+0]=(__bf16)x.x; f[4*v+1]=(__bf16)x.y; f[4*v+2]=(__bf16)x.z; f[4*v+3]=(__bf16)x.w;
  }
  return f;
}

// B-matrix fragment from a K-major (row index = k) matrix: B(k,n) = V[k*ldv + n].
__device__ __forceinline__ v16bf load_frag_b_kmajor(const float* V, int ldv,
                                                    int k_base, int n_base, int lane) {
  int n   = n_base + (lane & 15);
  int kb2 = k_base + ((lane >> 4) & 1) * 16;
  const float* p = V + (size_t)kb2 * ldv + n;
  v16bf f;
  #pragma unroll
  for (int e = 0; e < 16; ++e) f[e] = (__bf16)p[(size_t)e * ldv];
  return f;
}

__device__ __forceinline__ float gelu_tanh(float x) {
  float x3 = x * x * x;
  return 0.5f * x * (1.0f + tanhf(0.7978845608028654f * (x + 0.044715f * x3)));
}

// ---------------------------------------------------------------------------
// 1) elementwise silu for temb
// ---------------------------------------------------------------------------
__global__ __launch_bounds__(256) void silu_kernel(const float* __restrict__ x,
                                                   float* __restrict__ y, int n) {
  int i = blockIdx.x * blockDim.x + threadIdx.x;
  if (i < n) { float v = x[i]; y[i] = v / (1.0f + __expf(-v)); }
}

// ---------------------------------------------------------------------------
// 2) modulation GEMV: out[b,n] = dot(t[b,:], W[n,:]) + bias[n]   (M=B=2: no WMMA)
// ---------------------------------------------------------------------------
__global__ __launch_bounds__(256) void mod_gemv_kernel(const float* __restrict__ t,
                                                       const float* __restrict__ W,
                                                       const float* __restrict__ bias,
                                                       float* __restrict__ out) {
  __shared__ float red[256];
  int n = blockIdx.x, b = blockIdx.y, tid = threadIdx.x;
  const float* tb = t + (size_t)b * DD;
  const float* wr = W + (size_t)n * DD;
  float s = 0.f;
  for (int k = tid; k < DD; k += 256) s += tb[k] * wr[k];
  red[tid] = s; __syncthreads();
  for (int off = 128; off > 0; off >>= 1) {
    if (tid < off) red[tid] += red[tid + off];
    __syncthreads();
  }
  if (tid == 0) out[(size_t)b * 6 * DD + n] = red[0] + bias[n];
}

// ---------------------------------------------------------------------------
// 3) LayerNorm (no affine) + modulate: y = ln(x)*(1+scale) + shift
//    mod: [B, 6D]; shift at mod_off, scale at mod_off + D
// ---------------------------------------------------------------------------
__global__ __launch_bounds__(256) void ln_mod_kernel(const float* __restrict__ X,
                                                     const float* __restrict__ mod,
                                                     int mod_off,
                                                     float* __restrict__ Y,
                                                     int rows_per_batch) {
  __shared__ float s1[256], s2[256];
  int row = blockIdx.x, tid = threadIdx.x;
  int b = row / rows_per_batch;
  const float* x = X + (size_t)row * DD;
  float sum = 0.f, sq = 0.f;
  for (int c = tid; c < DD; c += 256) { float v = x[c]; sum += v; sq += v * v; }
  s1[tid] = sum; s2[tid] = sq; __syncthreads();
  for (int off = 128; off > 0; off >>= 1) {
    if (tid < off) { s1[tid] += s1[tid + off]; s2[tid] += s2[tid + off]; }
    __syncthreads();
  }
  float mean = s1[0] * (1.0f / DD);
  float var  = s2[0] * (1.0f / DD) - mean * mean;
  float inv  = rsqrtf(var + EPS);
  const float* mb = mod + (size_t)b * 6 * DD + mod_off;
  float* y = Y + (size_t)row * DD;
  for (int c = tid; c < DD; c += 256) {
    float shift = mb[c], scale = mb[DD + c];
    y[c] = (x[c] - mean) * inv * (1.0f + scale) + shift;
  }
}

// ---------------------------------------------------------------------------
// 4) WMMA GEMM: C[M,N] = A[M,K] @ W[N,K]^T + bias
//    mode 0: plain ; mode 1: tanh-GELU ; mode 2: out = resid + gate * (.)
//    Block 256 thr = 8 waves (2 M x 4 N); wave tile 64x32; block tile 128x128.
//    All launches use M,N multiples of 128 and K multiple of 32.
// ---------------------------------------------------------------------------
__global__ __launch_bounds__(256) void gemm_kernel(const float* __restrict__ A, int lda,
                                                   const float* __restrict__ W, int K,
                                                   const float* __restrict__ bias,
                                                   float* __restrict__ C, int ldc,
                                                   int M, int N, int mode,
                                                   const float* __restrict__ resid,
                                                   const float* __restrict__ gate) {
  int tid = threadIdx.x, lane = tid & 31, w = tid >> 5;
  int wm = w & 1, wn = w >> 1;
  int m0 = blockIdx.y * 128 + wm * 64;
  int n0 = blockIdx.x * 128 + wn * 32;
  if (m0 >= M || n0 >= N) return;

  v8f acc[4][2] = {};
  for (int kk = 0; kk < K; kk += 32) {
    if (kk + 64 <= K)  // pull next W chunk toward L2 (global_prefetch_b8)
      __builtin_prefetch(W + (size_t)(n0 + (lane & 15)) * K + kk + 32, 0, 1);
    v16bf b0 = load_frag_b(W, K, n0,      kk, lane);
    v16bf b1 = load_frag_b(W, K, n0 + 16, kk, lane);
    #pragma unroll
    for (int t = 0; t < 4; ++t) {
      v16bf a = load_frag_a(A, lda, m0 + t * 16, kk, lane);
      acc[t][0] = wmma_bf16(a, b0, acc[t][0]);
      acc[t][1] = wmma_bf16(a, b1, acc[t][1]);
    }
  }

  // epilogue: D tile layout -> lane(0-15)=N, half-wave selects M rows 0-7 / 8-15
  int nc = lane & 15;
  int mb = ((lane >> 4) & 1) * 8;
  #pragma unroll
  for (int t = 0; t < 4; ++t) {
    #pragma unroll
    for (int j = 0; j < 2; ++j) {
      int col = n0 + j * 16 + nc;
      float bv = bias ? bias[col] : 0.f;
      #pragma unroll
      for (int r = 0; r < 8; ++r) {
        int row = m0 + t * 16 + mb + r;
        float v = acc[t][j][r] + bv;
        if (mode == 1) v = gelu_tanh(v);
        else if (mode == 2) v = resid[(size_t)row * ldc + col] + gate[col] * v;
        C[(size_t)row * ldc + col] = v;
      }
    }
  }
}

// ---------------------------------------------------------------------------
// 5) per-head RMSNorm + RoPE + txt-first concat repack:
//    dst[b,h,s,dh], s<S_TXT from txt stream, else img stream.
// ---------------------------------------------------------------------------
__global__ __launch_bounds__(128) void qkv_pack_kernel(const float* __restrict__ img_src,
                                                       const float* __restrict__ txt_src,
                                                       const float* __restrict__ w_img,
                                                       const float* __restrict__ w_txt,
                                                       const float* __restrict__ freqs,
                                                       float* __restrict__ dst,
                                                       int do_rms_rope) {
  __shared__ float sh[128];
  __shared__ float pr[128];
  int s = blockIdx.x, h = blockIdx.y, b = blockIdx.z, d = threadIdx.x;
  const float* src = (s < S_TXT)
      ? txt_src + ((size_t)(b * S_TXT + s) * DD + h * DH)
      : img_src + ((size_t)(b * S_IMG + (s - S_TXT)) * DD + h * DH);
  float x = src[d];
  if (do_rms_rope) {
    sh[d] = x * x; __syncthreads();
    for (int off = 64; off > 0; off >>= 1) {
      if (d < off) sh[d] += sh[d + off];
      __syncthreads();
    }
    float rms = rsqrtf(sh[0] * (1.0f / DH) + EPS);
    const float* wgt = (s < S_TXT) ? w_txt : w_img;
    x = x * rms * wgt[d];
    pr[d] = x; __syncthreads();
    int p = d >> 1, j = d & 1;
    const float* f = freqs + ((size_t)s * (DH / 2) + p) * 4 + j * 2;
    x = f[0] * pr[2 * p] + f[1] * pr[2 * p + 1];
  }
  dst[(((size_t)b * HH + h) * SS + s) * DH + d] = x;
}

// ---------------------------------------------------------------------------
// 6) WMMA attention. One block per (b,h, 32-row q tile). LDS holds the full
//    32 x 1280 fp32 score strip (160KB) -> exact two-pass softmax, then PV.
// ---------------------------------------------------------------------------
#define ATTN_LDS_FLOATS (32 * SS + 256 + 32)

__global__ __launch_bounds__(256) void attn_kernel(const float* __restrict__ Q,
                                                   const float* __restrict__ K,
                                                   const float* __restrict__ V,
                                                   float* __restrict__ O,
                                                   float scale) {
  extern __shared__ float lds[];
  float* sc     = lds;                // [32][SS]
  float* red    = lds + 32 * SS;      // [32][8]
  float* rowsum = red + 256;          // [32]

  int qb = blockIdx.x, h = blockIdx.y, b = blockIdx.z;
  int tid = threadIdx.x, lane = tid & 31, w = tid >> 5;
  const float* Qb = Q + ((size_t)(b * HH + h) * SS) * DH;
  const float* Kb = K + ((size_t)(b * HH + h) * SS) * DH;
  const float* Vb = V + ((size_t)(b * HH + h) * SS) * DH;

  int nc = lane & 15;
  int mb = ((lane >> 4) & 1) * 8;

  // ---- phase 1: S = (Q Ktile^T) * scale, streamed into LDS -----------------
  v16bf qa[2][4];
  #pragma unroll
  for (int c = 0; c < 4; ++c) {
    qa[0][c] = load_frag_a(Qb, DH, qb * 32,      c * 32, lane);
    qa[1][c] = load_frag_a(Qb, DH, qb * 32 + 16, c * 32, lane);
  }
  for (int kb = w; kb < SS / 16; kb += 8) {
    v8f s0 = {}, s1 = {};
    #pragma unroll
    for (int c = 0; c < 4; ++c) {
      v16bf bf = load_frag_b(Kb, DH, kb * 16, c * 32, lane);  // keys are W rows
      s0 = wmma_bf16(qa[0][c], bf, s0);
      s1 = wmma_bf16(qa[1][c], bf, s1);
    }
    #pragma unroll
    for (int r = 0; r < 8; ++r) {
      sc[(mb + r) * SS + kb * 16 + nc]        = s0[r] * scale;
      sc[(16 + mb + r) * SS + kb * 16 + nc]   = s1[r] * scale;
    }
  }
  __syncthreads();

  // ---- phase 2: exact softmax over each of the 32 rows (8 threads/row) -----
  int row = tid >> 3, sub = tid & 7;
  float mx = -3.0e38f;
  for (int c = sub; c < SS; c += 8) mx = fmaxf(mx, sc[row * SS + c]);
  red[row * 8 + sub] = mx; __syncthreads();
  mx = red[row * 8];
  #pragma unroll
  for (int i = 1; i < 8; ++i) mx = fmaxf(mx, red[row * 8 + i]);
  __syncthreads();
  float sum = 0.f;
  for (int c = sub; c < SS; c += 8) {
    float e = __expf(sc[row * SS + c] - mx);
    sc[row * SS + c] = e;
    sum += e;
  }
  red[row * 8 + sub] = sum; __syncthreads();
  if (sub == 0) {
    float t = 0.f;
    #pragma unroll
    for (int i = 0; i < 8; ++i) t += red[row * 8 + i];
    rowsum[row] = 1.0f / t;
  }
  __syncthreads();

  // ---- phase 3: O = P V  (A from LDS, B k-major from global) ---------------
  int n0 = w * 16;                       // each wave owns 16 of the 128 dh cols
  v8f o0 = {}, o1 = {};
  for (int kk = 0; kk < SS; kk += 32) {
    v16bf bf = load_frag_b_kmajor(Vb, DH, kk, n0, lane);
    v16bf a0 = load_frag_a(sc, SS, 0,  kk, lane);
    v16bf a1 = load_frag_a(sc, SS, 16, kk, lane);
    o0 = wmma_bf16(a0, bf, o0);
    o1 = wmma_bf16(a1, bf, o1);
  }
  #pragma unroll
  for (int r = 0; r < 8; ++r) {
    int m0r = mb + r, m1r = 16 + mb + r;
    int s0g = qb * 32 + m0r, s1g = qb * 32 + m1r;
    O[(((size_t)b * SS + s0g) * HH + h) * DH + n0 + nc] = o0[r] * rowsum[m0r];
    O[(((size_t)b * SS + s1g) * HH + h) * DH + n0 + nc] = o1[r] * rowsum[m1r];
  }
}

// ---------------------------------------------------------------------------
// Host orchestration
// ---------------------------------------------------------------------------
extern "C" void kernel_launch(void* const* d_in, const int* in_sizes, int n_in,
                              void* d_out, int out_size, void* d_ws, size_t ws_size,
                              hipStream_t stream) {
  const float* hidden   = (const float*)d_in[0];   // [B,S_IMG,D]
  const float* enc      = (const float*)d_in[1];   // [B,S_TXT,D]
  const float* temb     = (const float*)d_in[2];   // [B,D]
  const float* freqs    = (const float*)d_in[3];   // [1,S,1,64,2,2]
  const float* img_mod_w= (const float*)d_in[4];
  const float* img_mod_b= (const float*)d_in[5];
  const float* txt_mod_w= (const float*)d_in[6];
  const float* txt_mod_b= (const float*)d_in[7];
  const float* wq = (const float*)d_in[8];  const float* bq = (const float*)d_in[9];
  const float* wk = (const float*)d_in[10]; const float* bk = (const float*)d_in[11];
  const float* wv = (const float*)d_in[12]; const float* bv = (const float*)d_in[13];
  const float* awq= (const float*)d_in[14]; const float* abq= (const float*)d_in[15];
  const float* awk= (const float*)d_in[16]; const float* abk= (const float*)d_in[17];
  const float* awv= (const float*)d_in[18]; const float* abv= (const float*)d_in[19];
  const float* nq_w = (const float*)d_in[20]; const float* nk_w = (const float*)d_in[21];
  const float* naq_w= (const float*)d_in[22]; const float* nak_w= (const float*)d_in[23];
  const float* wo = (const float*)d_in[24]; const float* bo = (const float*)d_in[25];
  const float* wao= (const float*)d_in[26]; const float* bao= (const float*)d_in[27];
  const float* mw1= (const float*)d_in[28]; const float* mb1= (const float*)d_in[29];
  const float* mw2= (const float*)d_in[30]; const float* mb2= (const float*)d_in[31];
  const float* tw1= (const float*)d_in[32]; const float* tb1= (const float*)d_in[33];
  const float* tw2= (const float*)d_in[34]; const float* tb2= (const float*)d_in[35];
  float* out = (float*)d_out;

  // workspace layout (fp32), ~410 MB total
  const size_t NI = (size_t)BB * S_IMG * DD;   // img-stream activation
  const size_t NT = (size_t)BB * S_TXT * DD;   // txt-stream activation
  const size_t NS = (size_t)BB * SS * DD;      // concat activation
  float* ws      = (float*)d_ws;
  float* t_silu  = ws;                  // B*D
  float* img_mod = t_silu + BB * DD;    // B*6D
  float* txt_mod = img_mod + BB * 6 * DD;
  float* img_m   = txt_mod + BB * 6 * DD;     // NI (reused as im2 later)
  float* txt_m   = img_m + NI;                // NT (reused as tm2)
  float* iq = txt_m + NT; float* ik = iq + NI; float* iv = ik + NI;
  float* tq = iv + NI;    float* tk = tq + NT; float* tv = tk + NT;
  float* Qb = tv + NT; float* Kb = Qb + NS; float* Vb = Kb + NS;
  float* Ob = Vb + NS;                        // [B,S,H*DH] (already /rowsum)
  float* hs  = Ob + NS;                       // NI  post-attn img residual
  float* ehs = hs + NI;                       // NT  post-attn txt residual
  float* ffi = ehs + NT;                      // B*S_IMG*FF
  float* fft = ffi + (size_t)BB * S_IMG * FF; // B*S_TXT*FF

  // 1) t = silu(temb)
  silu_kernel<<<(BB * DD + 255) / 256, 256, 0, stream>>>(temb, t_silu, BB * DD);

  // 2) modulation vectors [B, 6D]
  mod_gemv_kernel<<<dim3(6 * DD, BB), 256, 0, stream>>>(t_silu, img_mod_w, img_mod_b, img_mod);
  mod_gemv_kernel<<<dim3(6 * DD, BB), 256, 0, stream>>>(t_silu, txt_mod_w, txt_mod_b, txt_mod);

  // 3) LN + modulate (mod1: shift@0, scale@D)
  ln_mod_kernel<<<BB * S_IMG, 256, 0, stream>>>(hidden, img_mod, 0, img_m, S_IMG);
  ln_mod_kernel<<<BB * S_TXT, 256, 0, stream>>>(enc,    txt_mod, 0, txt_m, S_TXT);

  // 4) QKV projections (WMMA)
  dim3 gI(DD / 128, (BB * S_IMG) / 128), gT(DD / 128, (BB * S_TXT) / 128);
  gemm_kernel<<<gI, 256, 0, stream>>>(img_m, DD, wq,  DD, bq,  iq, DD, BB * S_IMG, DD, 0, nullptr, nullptr);
  gemm_kernel<<<gI, 256, 0, stream>>>(img_m, DD, wk,  DD, bk,  ik, DD, BB * S_IMG, DD, 0, nullptr, nullptr);
  gemm_kernel<<<gI, 256, 0, stream>>>(img_m, DD, wv,  DD, bv,  iv, DD, BB * S_IMG, DD, 0, nullptr, nullptr);
  gemm_kernel<<<gT, 256, 0, stream>>>(txt_m, DD, awq, DD, abq, tq, DD, BB * S_TXT, DD, 0, nullptr, nullptr);
  gemm_kernel<<<gT, 256, 0, stream>>>(txt_m, DD, awk, DD, abk, tk, DD, BB * S_TXT, DD, 0, nullptr, nullptr);
  gemm_kernel<<<gT, 256, 0, stream>>>(txt_m, DD, awv, DD, abv, tv, DD, BB * S_TXT, DD, 0, nullptr, nullptr);

  // 5) RMSNorm + RoPE + txt-first concat into [B,H,S,DH]
  dim3 gP(SS, HH, BB);
  qkv_pack_kernel<<<gP, 128, 0, stream>>>(iq, tq, nq_w, naq_w, freqs, Qb, 1);
  qkv_pack_kernel<<<gP, 128, 0, stream>>>(ik, tk, nk_w, nak_w, freqs, Kb, 1);
  qkv_pack_kernel<<<gP, 128, 0, stream>>>(iv, tv, nullptr, nullptr, freqs, Vb, 0);

  // 6) attention -> Ob [B,S,H*DH]
  attn_kernel<<<dim3(SS / 32, HH, BB), 256, ATTN_LDS_FLOATS * sizeof(float), stream>>>(
      Qb, Kb, Vb, Ob, 0.08838834764831845f /* 1/sqrt(128) */);

  // 7) out-projections fused with gate1 residual (per batch: O rows not contiguous)
  for (int b = 0; b < BB; ++b) {
    const float* gi = img_mod + (size_t)b * 6 * DD + 2 * DD;  // img gate1
    const float* gt = txt_mod + (size_t)b * 6 * DD + 2 * DD;  // txt gate1
    gemm_kernel<<<dim3(DD / 128, S_TXT / 128), 256, 0, stream>>>(
        Ob + (size_t)b * SS * DD, DD, wao, DD, bao,
        ehs + (size_t)b * S_TXT * DD, DD, S_TXT, DD, 2,
        enc + (size_t)b * S_TXT * DD, gt);
    gemm_kernel<<<dim3(DD / 128, S_IMG / 128), 256, 0, stream>>>(
        Ob + ((size_t)b * SS + S_TXT) * DD, DD, wo, DD, bo,
        hs + (size_t)b * S_IMG * DD, DD, S_IMG, DD, 2,
        hidden + (size_t)b * S_IMG * DD, gi);
  }

  // 8) LN + modulate for MLP (mod2: shift@3D, scale@4D) — reuse img_m/txt_m
  ln_mod_kernel<<<BB * S_IMG, 256, 0, stream>>>(hs,  img_mod, 3 * DD, img_m, S_IMG);
  ln_mod_kernel<<<BB * S_TXT, 256, 0, stream>>>(ehs, txt_mod, 3 * DD, txt_m, S_TXT);

  // 9) MLP up-proj with fused tanh-GELU
  gemm_kernel<<<dim3(FF / 128, (BB * S_IMG) / 128), 256, 0, stream>>>(
      img_m, DD, mw1, DD, mb1, ffi, FF, BB * S_IMG, FF, 1, nullptr, nullptr);
  gemm_kernel<<<dim3(FF / 128, (BB * S_TXT) / 128), 256, 0, stream>>>(
      txt_m, DD, tw1, DD, tb1, fft, FF, BB * S_TXT, FF, 1, nullptr, nullptr);

  // 10) MLP down-proj fused with gate2 residual, writing straight into d_out
  //     d_out layout: [B, S_TXT + S_IMG, D], txt rows first per batch.
  for (int b = 0; b < BB; ++b) {
    const float* gi = img_mod + (size_t)b * 6 * DD + 5 * DD;  // img gate2
    const float* gt = txt_mod + (size_t)b * 6 * DD + 5 * DD;  // txt gate2
    gemm_kernel<<<dim3(DD / 128, S_TXT / 128), 256, 0, stream>>>(
        fft + (size_t)b * S_TXT * FF, FF, tw2, FF, tb2,
        out + (size_t)b * SS * DD, DD, S_TXT, DD, 2,
        ehs + (size_t)b * S_TXT * DD, gt);
    gemm_kernel<<<dim3(DD / 128, S_IMG / 128), 256, 0, stream>>>(
        ffi + (size_t)b * S_IMG * FF, FF, mw2, FF, mb2,
        out + ((size_t)b * SS + S_TXT) * DD, DD, S_IMG, DD, 2,
        hs + (size_t)b * S_IMG * DD, gi);
  }
}
